// BayesianLinear_36644660969989
// MI455X (gfx1250) — matrix operand order repaired and checked by reading, not compiled, not run on hardware
//
#include <hip/hip_runtime.h>
#include <hip/hip_bf16.h>

typedef __attribute__((ext_vector_type(16))) _Float16 v16h;
typedef __attribute__((ext_vector_type(8)))  _Float16 v8h;
typedef __attribute__((ext_vector_type(4)))  _Float16 v4h;
typedef __attribute__((ext_vector_type(8)))  float    v8f;

#define B_DIM   8192
#define IN_DIM  4096
#define OUT_DIM 4096
#define KP1     4097   // IN+1
#define LDT     40     // LDS row pitch in halves (32 + 8 pad, conflict-free)

// ---------------------------------------------------------------------------
// Kernel 1: sample weights (f32 -> f16), extract bias column, per-block KL sums
// One block per output row o; threads stride the 4097 columns.
// ---------------------------------------------------------------------------
__global__ __launch_bounds__(256)
void bayes_prep_kernel(const float* __restrict__ q_mu,
                       const float* __restrict__ q_ls,
                       const float* __restrict__ p_mu,
                       const float* __restrict__ eps,
                       const float* __restrict__ p_ls_ptr,
                       _Float16*    __restrict__ Wh,     // [OUT][IN] f16
                       float*       __restrict__ bias,   // [OUT] f32
                       float*       __restrict__ partials)
{
    const int o   = blockIdx.x;
    const int tid = threadIdx.x;
    const long long base = (long long)o * KP1;

    const float pls     = p_ls_ptr[0];
    const float inv2ps2 = 0.5f * __expf(-2.0f * pls);   // 1/(2*p_sigma^2)

    float lsum = 0.0f;
    for (int c = tid; c < KP1; c += 256) {
        const long long i = base + c;
        const float qm = q_mu[i];
        const float ql = q_ls[i];
        const float pm = p_mu[i];
        const float ev = eps[i];
        const float qs = __expf(ql);
        const float w  = qm + qs * ev;
        if (c < IN_DIM)
            Wh[(long long)o * IN_DIM + c] = (_Float16)w;
        else
            bias[o] = w;
        const float d = pm - qm;
        lsum += (pls - ql) + (qs * qs + d * d) * inv2ps2 - 0.5f;
    }

    __shared__ float red[256];
    red[tid] = lsum;
    __syncthreads();
    #pragma unroll
    for (int s = 128; s > 0; s >>= 1) {
        if (tid < s) red[tid] += red[tid + s];
        __syncthreads();
    }
    if (tid == 0) partials[o] = red[0];
}

// ---------------------------------------------------------------------------
// Kernel 2: deterministic final KL reduction (single block, fixed order)
// ---------------------------------------------------------------------------
__global__ __launch_bounds__(256)
void bayes_kl_finalize(const float* __restrict__ partials, int n,
                       float* __restrict__ out_kl)
{
    const int tid = threadIdx.x;
    float s = 0.0f;
    for (int i = tid; i < n; i += 256) s += partials[i];
    __shared__ float red[256];
    red[tid] = s;
    __syncthreads();
    #pragma unroll
    for (int k = 128; k > 0; k >>= 1) {
        if (tid < k) red[tid] += red[tid + k];
        __syncthreads();
    }
    if (tid == 0) out_kl[0] = red[0];
}

// ---------------------------------------------------------------------------
// Kernel 3: out[b][o] = sum_k x[b][k] * W[o][k] + bias[o]
// Block tile 256x128, BK=32, 256 threads = 8 wave32s in a 4(M) x 2(N) grid.
// Each wave computes a 64x64 tile = 4x4 v_wmma_f32_16x16x32_f16 accumulators
// (16 WMMAs per 16 ds_load_b128 per K-step). Double-buffered LDS: one barrier
// per K-step; tile k+1 is prefetched from global (L2-resident) into registers
// before computing tile k. x converted f32->f16 during staging.
// ---------------------------------------------------------------------------
__global__ __launch_bounds__(256)
void bayes_gemm_kernel(const float*    __restrict__ X,    // [B][IN] f32
                       const _Float16* __restrict__ Wh,   // [OUT][IN] f16
                       const float*    __restrict__ bias, // [OUT]
                       float*          __restrict__ Out)  // [B][OUT]
{
    __shared__ _Float16 As[2][256 * LDT];   // 2 x 20 KB
    __shared__ _Float16 Bs[2][128 * LDT];   // 2 x 10 KB

    const int tid    = threadIdx.x;
    const int lane   = tid & 31;
    const int wave   = tid >> 5;
    const int waveM  = wave >> 1;       // 0..3  -> 64-row slabs of 256
    const int waveN  = wave & 1;        // 0..1  -> 64-col slabs of 128
    const int laneLo = lane & 15;
    const int laneHi = lane >> 4;

    const int blockM = blockIdx.y * 256;
    const int blockN = blockIdx.x * 128;

    v8f acc[4][4];
    #pragma unroll
    for (int mt = 0; mt < 4; ++mt)
        #pragma unroll
        for (int nt = 0; nt < 4; ++nt)
            acc[mt][nt] = (v8f){0.f,0.f,0.f,0.f,0.f,0.f,0.f,0.f};

    // Staging coordinates: A = 256 rows x 32 f32 (2048 float4 chunks),
    //                      B = 128 rows x 32 f16 (512 v8h chunks)
    int aRow[8], aCol[8];
    #pragma unroll
    for (int i = 0; i < 8; ++i) {
        const int c = tid + i * 256;
        aRow[i] = c >> 3;
        aCol[i] = (c & 7) * 4;
    }
    int bRow[2], bCol[2];
    #pragma unroll
    for (int i = 0; i < 2; ++i) {
        const int c = tid + i * 256;
        bRow[i] = c >> 2;
        bCol[i] = (c & 3) * 8;
    }

    float4 aReg[8];
    v8h    bReg[2];

    // ---- preload K-step 0 and stage into buffer 0 ----
    #pragma unroll
    for (int i = 0; i < 8; ++i)
        aReg[i] = *(const float4*)(X + (size_t)(blockM + aRow[i]) * IN_DIM + aCol[i]);
    #pragma unroll
    for (int i = 0; i < 2; ++i)
        bReg[i] = *(const v8h*)(Wh + (size_t)(blockN + bRow[i]) * IN_DIM + bCol[i]);

    #pragma unroll
    for (int i = 0; i < 8; ++i) {
        v4h h = { (_Float16)aReg[i].x, (_Float16)aReg[i].y,
                  (_Float16)aReg[i].z, (_Float16)aReg[i].w };
        *(v4h*)(&As[0][aRow[i] * LDT + aCol[i]]) = h;
    }
    #pragma unroll
    for (int i = 0; i < 2; ++i)
        *(v8h*)(&Bs[0][bRow[i] * LDT + bCol[i]]) = bReg[i];

    int p = 0;
    for (int k0 = 0; k0 < IN_DIM; k0 += 32) {
        __syncthreads();   // buf[p] visible; everyone done reading buf[1-p]

        const bool more = (k0 + 32 < IN_DIM);
        if (more) {        // prefetch next tile from global (hides under WMMA)
            const int kn = k0 + 32;
            #pragma unroll
            for (int i = 0; i < 8; ++i)
                aReg[i] = *(const float4*)(X + (size_t)(blockM + aRow[i]) * IN_DIM
                                             + kn + aCol[i]);
            #pragma unroll
            for (int i = 0; i < 2; ++i)
                bReg[i] = *(const v8h*)(Wh + (size_t)(blockN + bRow[i]) * IN_DIM
                                           + kn + bCol[i]);
        }

        // ---- compute one K=32 step from buf[p] ----
        // A frag (16-bit A 16x32): lanes 0-15: M=lane, K={0..7,16..23};
        //                          lanes 16-31: M=lane-16, K={8..15,24..31}
        v16h af[4];
        #pragma unroll
        for (int mt = 0; mt < 4; ++mt) {
            const int r = waveM * 64 + mt * 16 + laneLo;
            const _Float16* ap = &As[p][r * LDT + laneHi * 8];
            ((v8h*)&af[mt])[0] = *(const v8h*)(ap);        // ds_load_b128
            ((v8h*)&af[mt])[1] = *(const v8h*)(ap + 16);
        }
        // B frag (16-bit B 32x16): lanes 0-15: N=lane, K={0..7,16..23};
        //                          lanes 16-31: N=lane-16, K={8..15,24..31}
        #pragma unroll
        for (int nt = 0; nt < 4; ++nt) {
            const int r = waveN * 64 + nt * 16 + laneLo;
            const _Float16* bp = &Bs[p][r * LDT + laneHi * 8];
            v16h bf;
            ((v8h*)&bf)[0] = *(const v8h*)(bp);
            ((v8h*)&bf)[1] = *(const v8h*)(bp + 16);
            #pragma unroll
            for (int mt = 0; mt < 4; ++mt)
                acc[mt][nt] = __builtin_amdgcn_wmma_f32_16x16x32_f16(
                    false, af[mt], false, bf,
                    (short)0, acc[mt][nt], false, false);
        }

        // ---- stage prefetched tile into the other buffer ----
        if (more) {
            #pragma unroll
            for (int i = 0; i < 8; ++i) {
                v4h h = { (_Float16)aReg[i].x, (_Float16)aReg[i].y,
                          (_Float16)aReg[i].z, (_Float16)aReg[i].w };
                *(v4h*)(&As[1 - p][aRow[i] * LDT + aCol[i]]) = h;
            }
            #pragma unroll
            for (int i = 0; i < 2; ++i)
                *(v8h*)(&Bs[1 - p][bRow[i] * LDT + bCol[i]]) = bReg[i];
        }
        p ^= 1;
    }

    // ---- epilogue: fused bias add, direct f32 stores ----
    // C/D layout: VGPR r -> M = r (lanes 0-15) / r+8 (lanes 16-31), N = lane&15
    #pragma unroll
    for (int nt = 0; nt < 4; ++nt) {
        const int n   = blockN + waveN * 64 + nt * 16 + laneLo;
        const float bv = bias[n];
        const int rowBase = blockM + waveM * 64 + laneHi * 8;
        #pragma unroll
        for (int mt = 0; mt < 4; ++mt)
            #pragma unroll
            for (int r = 0; r < 8; ++r)
                Out[(size_t)(rowBase + mt * 16 + r) * OUT_DIM + n]
                    = acc[mt][nt][r] + bv;
    }
}

// ---------------------------------------------------------------------------
extern "C" void kernel_launch(void* const* d_in, const int* in_sizes, int n_in,
                              void* d_out, int out_size, void* d_ws, size_t ws_size,
                              hipStream_t stream) {
    const float* x    = (const float*)d_in[0];
    const float* pls  = (const float*)d_in[1];
    const float* q_mu = (const float*)d_in[2];
    const float* q_ls = (const float*)d_in[3];
    const float* p_mu = (const float*)d_in[4];
    const float* eps  = (const float*)d_in[5];
    float* out = (float*)d_out;

    // Workspace layout:
    //   Wh       : f16 [4096][4096]  = 33,554,432 B
    //   bias     : f32 [4096]        =     16,384 B
    //   partials : f32 [4096]        =     16,384 B
    char* ws = (char*)d_ws;
    _Float16* Wh       = (_Float16*)ws;
    float*    bias     = (float*)(ws + 33554432);
    float*    partials = (float*)(ws + 33554432 + 16384);

    bayes_prep_kernel<<<OUT_DIM, 256, 0, stream>>>(q_mu, q_ls, p_mu, eps, pls,
                                                   Wh, bias, partials);
    bayes_kl_finalize<<<1, 256, 0, stream>>>(partials, OUT_DIM,
                                             out + (size_t)B_DIM * OUT_DIM);
    dim3 grid(OUT_DIM / 128, B_DIM / 256);  // (32, 32)
    bayes_gemm_kernel<<<grid, 256, 0, stream>>>(x, Wh, bias, out);
}
